// VectorQuantizer_36550171689545
// MI455X (gfx1250) — matrix operand order repaired and checked by reading, not compiled, not run on hardware
//
#include <hip/hip_runtime.h>
#include <hip/hip_bf16.h>

typedef _Float16 v16h __attribute__((ext_vector_type(16)));
typedef _Float16 v8h  __attribute__((ext_vector_type(8)));
typedef float    v8f  __attribute__((ext_vector_type(8)));

#define Bn   8
#define Dn   1024
#define Tn   4096
#define CDn  8
#define CSn  1024
#define TT   64              // timesteps per block
#define NTILE (Tn / TT)      // 64 tiles along T
#define THREADS 256
#define DCH (Dn / 4)         // 256 channels per quarter-block
#define PFD 64               // z prefetch distance in rows (64*16KB = 1MB offset)

// d_out flat layout (fp32 words; indices stored as int32 bit patterns)
#define COMMIT_OFF (Bn * Dn * Tn)          // 33,554,432
#define CBL_OFF    (COMMIT_OFF + Bn)
#define IDX_OFF    (CBL_OFF + Bn)
#define ZE_OFF     (IDX_OFF + Bn * Tn)

__global__ __launch_bounds__(THREADS) void vq_fused_kernel(
    const float* __restrict__ z,     const float* __restrict__ in_w,
    const float* __restrict__ in_b,  const float* __restrict__ out_w,
    const float* __restrict__ out_b, const float* __restrict__ cbook,
    float* __restrict__ outF,        float* __restrict__ partials)
{
  __shared__ __attribute__((aligned(16))) float ldsInW[Dn * CDn];   // [d][c] transposed
  __shared__ __attribute__((aligned(16))) float ldsOutW[Dn * CDn];  // [d][c] (async DMA)
  __shared__ __attribute__((aligned(16))) float ldsOutB[Dn];        // (async DMA)
  __shared__ v16h  cbA2[CSn];   // normalized codebook f16, zero-padded K=8..15 (32KB)
  __shared__ __attribute__((aligned(32))) _Float16 zeB[CDn][TT];    // z_e tile, f16
  __shared__ float accP[3][TT][CDn];                                // D-chunk partial sums
  __shared__ float bestS[4][16];                                    // cross-wave argmax merge
  __shared__ int   bestIx[4][16];
  __shared__ int   idxLds[TT];
  __shared__ float red[8];                                          // per-wave loss sums

  const int tid   = threadIdx.x;
  const int b     = blockIdx.y;
  const int t0    = blockIdx.x * TT;
  const int lane  = tid & 31;
  const int wave  = tid >> 5;
  const int tl    = tid & (TT - 1);      // timestep owned within tile
  const int chunk = tid >> 6;            // which quarter of D (0..3)
  const int t     = t0 + tl;

  // ---------- async DMA: stage out_w / out_b into LDS, overlapped with A/B -----
  // global_load_async_to_lds_b128: per-lane 16B copy, tracked by ASYNCcnt.
  {
    const unsigned dstW = (unsigned)(uintptr_t)(void*)&ldsOutW[0];
    #pragma unroll
    for (int i = tid; i < (Dn * CDn) / 4; i += THREADS) {           // 8 iters
      const float* src = out_w + (size_t)i * 4;
      unsigned dst = dstW + (unsigned)i * 16u;
      asm volatile("global_load_async_to_lds_b128 %0, %1, off"
                   :: "v"(dst), "v"(src) : "memory");
    }
    const unsigned dstB = (unsigned)(uintptr_t)(void*)&ldsOutB[0];
    {
      const float* src = out_b + (size_t)tid * 4;                   // 256 xfers
      unsigned dst = dstB + (unsigned)tid * 16u;
      asm volatile("global_load_async_to_lds_b128 %0, %1, off"
                   :: "v"(dst), "v"(src) : "memory");
    }
  }

  // ---------------- phase 0: stage in_w (transposed) + normalized codebook -----
  for (int i = tid; i < Dn * CDn; i += THREADS) {
    int c = i >> 10, d = i & (Dn - 1);
    ldsInW[d * CDn + c] = in_w[i];     // transpose [c][d] -> [d][c]
  }
  for (int c = tid; c < CSn; c += THREADS) {
    float v[CDn]; float ss = 0.f;
    #pragma unroll
    for (int k = 0; k < CDn; ++k) { v[k] = cbook[c * CDn + k]; ss = fmaf(v[k], v[k], ss); }
    float r = 1.f / fmaxf(sqrtf(ss), 1e-12f);
    v16h h = {};                       // K=8..15 stay exact zero in LDS
    #pragma unroll
    for (int k = 0; k < CDn; ++k) h[k] = (_Float16)(v[k] * r);
    cbA2[c] = h;
  }
  __syncthreads();

  // ---------------- phase A: z_e partials, D split 4 ways across waves ---------
  float acc[CDn];
  #pragma unroll
  for (int c = 0; c < CDn; ++c) acc[c] = 0.f;
  {
    const float* zp = z + ((size_t)b * Dn + (size_t)chunk * DCH) * (size_t)Tn + t;
    #pragma unroll 8
    for (int i = 0; i < DCH; ++i) {
      int d = chunk * DCH + i;
      float zv = __builtin_nontemporal_load(zp + (size_t)i * Tn);   // stream z once
      __builtin_prefetch(zp + (size_t)(i + PFD) * Tn, 0, 0);        // global_prefetch_b8
      const float4 w0 = *(const float4*)&ldsInW[d * CDn];           // LDS broadcast
      const float4 w1 = *(const float4*)&ldsInW[d * CDn + 4];
      acc[0] = fmaf(w0.x, zv, acc[0]); acc[1] = fmaf(w0.y, zv, acc[1]);
      acc[2] = fmaf(w0.z, zv, acc[2]); acc[3] = fmaf(w0.w, zv, acc[3]);
      acc[4] = fmaf(w1.x, zv, acc[4]); acc[5] = fmaf(w1.y, zv, acc[5]);
      acc[6] = fmaf(w1.z, zv, acc[6]); acc[7] = fmaf(w1.w, zv, acc[7]);
    }
  }
  if (chunk != 0) {
    #pragma unroll
    for (int c = 0; c < CDn; ++c) accP[chunk - 1][tl][c] = acc[c];
  }
  __syncthreads();

  float ze[CDn];
  #pragma unroll
  for (int c = 0; c < CDn; ++c) ze[c] = 0.f;
  if (chunk == 0) {
    #pragma unroll
    for (int c = 0; c < CDn; ++c) {
      ze[c] = acc[c] + accP[0][tl][c] + accP[1][tl][c] + accP[2][tl][c] + in_b[c];
      outF[ZE_OFF + ((size_t)b * CDn + c) * Tn + t] = ze[c];        // z_e output fp32
      zeB[c][tl] = (_Float16)ze[c];                                 // f16 copy for WMMA
    }
  }
  __syncthreads();

  // ---------------- phase B: argmax over 1024 codes via WMMA -------------------
  // wave w scores tile (w&3); waves 0-3 take codes 0..511, waves 4-7 take 512..1023
  const int tile   = wave & 3;
  const int ctBase = (wave >> 2) * 32;
  // B fragment: N=lane%16, element s holds K=s (s<8) for lanes 0-15. Exactly zero elsewhere.
  v16h bfr = {};
  if (lane < 16) {
    #pragma unroll
    for (int s = 0; s < CDn; ++s) bfr[s] = zeB[s][tile * 16 + lane];
  }

  // A fragment comes entirely from LDS (two ds_load_b128 straight into the
  // contiguous operand block). Lanes 16-31 duplicate rows; their K=8..15 slots
  // are annihilated by B==0 there.
  const int rowLane = lane & 15;
  float best = -3.0e38f; int bestI = 0;
  for (int ct = ctBase; ct < ctBase + 32; ct += 2) {   // manual 2x: pipeline ds loads
    v16h a0 = cbA2[ct * 16 + rowLane];
    v16h a1 = cbA2[(ct + 1) * 16 + rowLane];
    v8f c0 = {}, c1 = {};
    v8f d0 = __builtin_amdgcn_wmma_f32_16x16x32_f16(
        false, a0, false, bfr, (short)0, c0, false, false);
    v8f d1 = __builtin_amdgcn_wmma_f32_16x16x32_f16(
        false, a1, false, bfr, (short)0, c1, false, false);
    #pragma unroll
    for (int j = 0; j < 8; ++j) {
      int code = ct * 16 + j + ((lane >> 4) << 3);
      float s = d0[j];
      if (s > best) { best = s; bestI = code; }        // strict > keeps first max
    }
    #pragma unroll
    for (int j = 0; j < 8; ++j) {
      int code = (ct + 1) * 16 + j + ((lane >> 4) << 3);
      float s = d1[j];
      if (s > best) { best = s; bestI = code; }
    }
  }
  { // merge lane l with lane l+16 (same t column)
    float ob = __shfl_xor(best, 16, 32);
    int   oi = __shfl_xor(bestI, 16, 32);
    if (ob > best || (ob == best && oi < bestI)) { best = ob; bestI = oi; }
  }
  if (wave >= 4 && lane < 16) { bestS[tile][lane] = best; bestIx[tile][lane] = bestI; }
  __syncthreads();
  if (wave < 4 && lane < 16) {
    float hs = bestS[tile][lane]; int hi = bestIx[tile][lane];
    if (hs > best) { best = hs; bestI = hi; }          // ties -> low half = smaller index
    idxLds[tile * 16 + lane] = bestI;
    ((int*)outF)[IDX_OFF + b * Tn + t0 + tile * 16 + lane] = bestI;   // indices out
  }
  __syncthreads();
  const int myIdx = idxLds[tl];

  // ---------------- losses: mean((z_e - z_q)^2), deterministic reduction ------
  float zq[CDn];
  #pragma unroll
  for (int c = 0; c < CDn; ++c) zq[c] = cbook[myIdx * CDn + c];      // raw gather
  float ls = 0.f;
  if (chunk == 0) {
    #pragma unroll
    for (int c = 0; c < CDn; ++c) { float df = ze[c] - zq[c]; ls = fmaf(df, df, ls); }
  }
  #pragma unroll
  for (int off = 16; off > 0; off >>= 1) ls += __shfl_xor(ls, off, 32);  // fixed order
  if (lane == 0) red[wave] = ls;

  // fence the async out_w/out_b DMA: own-wave wait + barrier covers all waves
  asm volatile("s_wait_asynccnt 0" ::: "memory");
  __syncthreads();
  if (tid == 0) {
    float s = 0.f;
    #pragma unroll
    for (int w = 0; w < 8; ++w) s += red[w];                         // fixed order
    partials[b * NTILE + blockIdx.x] = s;
  }

  // ---------------- phase C: out = out_w @ z_q + out_b, D split 4 ways --------
  {
    float* op = outF + ((size_t)b * Dn + (size_t)chunk * DCH) * (size_t)Tn + t;
    #pragma unroll 4
    for (int i = 0; i < DCH; ++i) {
      int d = chunk * DCH + i;
      const float4 w0 = *(const float4*)&ldsOutW[d * CDn];
      const float4 w1 = *(const float4*)&ldsOutW[d * CDn + 4];
      float v = ldsOutB[d];
      v = fmaf(w0.x, zq[0], v); v = fmaf(w0.y, zq[1], v);
      v = fmaf(w0.z, zq[2], v); v = fmaf(w0.w, zq[3], v);
      v = fmaf(w1.x, zq[4], v); v = fmaf(w1.y, zq[5], v);
      v = fmaf(w1.z, zq[6], v); v = fmaf(w1.w, zq[7], v);
      __builtin_nontemporal_store(v, op + (size_t)i * Tn);           // stream out
    }
  }
}

// fixed-order reduction of per-block partials -> both losses (bit-deterministic)
__global__ void vq_finalize_kernel(const float* __restrict__ partials,
                                   float* __restrict__ outF)
{
  int b = threadIdx.x;
  if (b < Bn) {
    float s = 0.f;
    for (int i = 0; i < NTILE; ++i) s += partials[b * NTILE + i];
    float v = s / (float)(CDn * Tn);
    outF[COMMIT_OFF + b] = v;
    outF[CBL_OFF + b]    = v;     // numerically identical to commitment loss
  }
}

extern "C" void kernel_launch(void* const* d_in, const int* in_sizes, int n_in,
                              void* d_out, int out_size, void* d_ws, size_t ws_size,
                              hipStream_t stream) {
  const float* z     = (const float*)d_in[0];
  const float* in_w  = (const float*)d_in[1];
  const float* in_b  = (const float*)d_in[2];
  const float* out_w = (const float*)d_in[3];
  const float* out_b = (const float*)d_in[4];
  const float* cbook = (const float*)d_in[5];
  float* outF = (float*)d_out;
  float* part = (float*)d_ws;      // Bn*NTILE = 512 floats

  dim3 grid(NTILE, Bn);
  vq_fused_kernel<<<grid, THREADS, 0, stream>>>(z, in_w, in_b, out_w, out_b, cbook, outF, part);
  vq_finalize_kernel<<<1, 32, 0, stream>>>(part, outF);
}
